// SelfAttention_15951508537916
// MI455X (gfx1250) — matrix operand order repaired and checked by reading, not compiled, not run on hardware
//
#include <hip/hip_runtime.h>

// ---------------------------------------------------------------------------
// Llama-style self-attention block on gfx1250 (CDNA5, wave32, WMMA).
// B=2 S=2048 D=4096, 32 Q heads / 8 KV heads (GQA x4), head_dim=128.
// GEMMs: v_wmma_f32_16x16x32_bf16 fed from 64-wide-K LDS tiles staged by the
// Tensor Data Mover (tensor_load_to_lds + s_wait_tensorcnt), double buffered.
// ---------------------------------------------------------------------------

#define Bn   2
#define Sn   2048
#define Dn   4096
#define HQ   32
#define HKV  8
#define HD   128
#define MROW (Bn * Sn)          // 4096 GEMM rows
#define NQ   (HQ * HD)          // 4096
#define NKV  (HKV * HD)         // 1024

typedef __bf16 bf16;
typedef __attribute__((ext_vector_type(16))) __bf16       v16bf;
typedef __attribute__((ext_vector_type(8)))  float        v8f;
typedef __attribute__((ext_vector_type(4)))  unsigned int u32x4;
typedef __attribute__((ext_vector_type(4)))  int          i32x4;
typedef __attribute__((ext_vector_type(8)))  int          i32x8;

#if defined(__has_builtin)
#  if __has_builtin(__builtin_amdgcn_tensor_load_to_lds) && \
      __has_builtin(__builtin_amdgcn_s_wait_tensorcnt)
#    define USE_TDM 1
#  endif
#endif
#ifndef USE_TDM
#  define USE_TDM 0
#  warning "CDNA5 TDM builtins unavailable -- using cooperative-copy fallback"
#endif

union FragU { u32x4 q[2]; v16bf v; };

__device__ __forceinline__ bf16 f2bf(float f) {
  unsigned u = __float_as_uint(f);
  u += 0x7FFFu + ((u >> 16) & 1u);           // round-to-nearest-even
  unsigned short h = (unsigned short)(u >> 16);
  return __builtin_bit_cast(bf16, h);
}
__device__ __forceinline__ float bf2f(bf16 h) {
  unsigned short us = __builtin_bit_cast(unsigned short, h);
  return __uint_as_float(((unsigned)us) << 16);
}

// A fragment (16x32 bf16): lanes 0-15 hold K {0..7,16..23}; lanes 16-31 hold
// K {8..15,24..31} (CDNA5 ISA 7.12.2).  `win` = start of this row's 32-K window.
__device__ __forceinline__ v16bf load_frag_a(const bf16* win, int lane) {
  const int off = (lane & 16) ? 8 : 0;
  FragU f;
  f.q[0] = *(const u32x4*)(win + off);
  f.q[1] = *(const u32x4*)(win + off + 16);
  return f.v;
}
// B fragment (32x16 bf16): lanes 0-15 hold K {0..7,16..23}; lanes 16-31 hold K {8..15,24..31}.
__device__ __forceinline__ v16bf load_frag_b(const bf16* win, int lane) {
  const int off = (lane & 16) ? 8 : 0;
  FragU f;
  f.q[0] = *(const u32x4*)(win + off);
  f.q[1] = *(const u32x4*)(win + off + 16);
  return f.v;
}

__device__ __forceinline__ v8f wmma_bf16(v16bf a, v16bf b, v8f c) {
  return __builtin_amdgcn_wmma_f32_16x16x32_bf16(false, a, false, b,
                                                 (short)0, c, false, false);
}

#if USE_TDM
// LDS byte offset of a __shared__ object (generic -> AS(3) -> integer).
__device__ __forceinline__ unsigned lds_offset(const void* p) {
  return (unsigned)(size_t)(const __attribute__((address_space(3))) void*)p;
}
// TDM: copy a [rows x cols] bf16 tile (row stride = strideElems) to LDS.
// Descriptor per CDNA5 ISA ch.8: group0 {count,lds_addr,global_addr,type=2},
// group1 {data_size=2B, tensor_dim0/1, tile_dim0/1, dim0_stride}.
__device__ __forceinline__ void tdm_load_tile(unsigned lds_off, const bf16* g,
                                              unsigned rows, unsigned cols,
                                              unsigned strideElems) {
  unsigned long long ga = (unsigned long long)(size_t)g;
  u32x4 g0;
  g0[0] = 1u;                                            // count=1, user mode
  g0[1] = lds_off;                                       // lds_addr (bytes)
  g0[2] = (unsigned)ga;                                  // global_addr[31:0]
  g0[3] = (unsigned)((ga >> 32) & 0x01FFFFFFull) | 0x80000000u; // [56:32]|type=2
  i32x8 g1;
  g1[0] = 0x00010000;                                    // data_size = 2 bytes
  g1[1] = (int)(cols << 16);                             // tensor_dim0 = cols
  g1[2] = (int)(rows << 16);                             // tensor_dim1 = rows
  g1[3] = (int)(cols << 16);                             // tile_dim0 = cols
  g1[4] = (int)rows;                                     // tile_dim1 = rows
  g1[5] = (int)strideElems;                              // dim0_stride lo
  g1[6] = 0;
  g1[7] = 0;
  i32x4 z4 = {0, 0, 0, 0};
#if __clang_major__ >= 23
  i32x8 z8 = {0, 0, 0, 0, 0, 0, 0, 0};
  __builtin_amdgcn_tensor_load_to_lds(g0, g1, z4, z4, z8, 0);
#else
  __builtin_amdgcn_tensor_load_to_lds(g0, g1, z4, z4, 0);
#endif
}
__device__ __forceinline__ void wait_tensor0() {
  __builtin_amdgcn_s_wait_tensorcnt(0);
}
#endif

// ---------------------------------------------------------------------------
// Elementwise converters
// ---------------------------------------------------------------------------
__global__ void k_f32_to_bf16(const float* __restrict__ in,
                              bf16* __restrict__ out, size_t n) {
  size_t i = (size_t)blockIdx.x * blockDim.x + threadIdx.x;
  size_t st = (size_t)gridDim.x * blockDim.x;
  for (; i < n; i += st) out[i] = f2bf(in[i]);
}

// in[R][C] f32 -> out[C][R] bf16 (weights N-major so B frags are contiguous)
__global__ void k_transpose_to_bf16(const float* __restrict__ in,
                                    bf16* __restrict__ out, int R, int C) {
  size_t n = (size_t)R * C;
  size_t i = (size_t)blockIdx.x * blockDim.x + threadIdx.x;
  size_t st = (size_t)gridDim.x * blockDim.x;
  for (; i < n; i += st) {
    int r = (int)(i / C), c = (int)(i % C);
    out[(size_t)c * R + r] = f2bf(in[i]);
  }
}

// ---------------------------------------------------------------------------
// bf16 GEMM: C[M,N] = A[M,K] * Bt[N,K]^T.
// Block = 4 waves, 64x128 block tile, 32x64 per wave, K-step 64.
// Per stage: TDM-DMA A(64x64)+B(128x64) into LDS (each wave its slice,
// double buffered), then 16 WMMA per wave between barriers.
// ---------------------------------------------------------------------------
#define KSTEP      64
#define TILE_A_ELE (64 * KSTEP)              // 4096 elems
#define TILE_B_ELE (128 * KSTEP)             // 8192 elems
#define TILE_ELE   (TILE_A_ELE + TILE_B_ELE) // 12288 elems = 24 KB / buffer

template <int OUT_F32>
__global__ void __launch_bounds__(128)
k_gemm_bf16(const bf16* __restrict__ A, const bf16* __restrict__ Bt,
            void* __restrict__ Cout, int M, int N, int K) {
  __shared__ bf16 smem[2][TILE_ELE];         // 48 KB double buffer

  const int lane = threadIdx.x & 31;
  const int w    = threadIdx.x >> 5;
  const int m0   = blockIdx.y * 64;
  const int n0   = blockIdx.x * 128;
  const int wm   = (w >> 1) * 32;
  const int wn   = (w & 1) * 64;
  const int cn   = lane & 15;

#if USE_TDM
  const unsigned ldsA0 = lds_offset(&smem[0][0]);
  // stage buffer `buf` with the 64-K window starting at k0
  auto stage = [&](int buf, int k0) {
    unsigned base = ldsA0 + (unsigned)buf * (TILE_ELE * 2);
    tdm_load_tile(base + (unsigned)(w * 16) * (KSTEP * 2),
                  A + (size_t)(m0 + w * 16) * K + k0, 16, KSTEP, (unsigned)K);
    tdm_load_tile(base + TILE_A_ELE * 2 + (unsigned)(w * 32) * (KSTEP * 2),
                  Bt + (size_t)(n0 + w * 32) * K + k0, 32, KSTEP, (unsigned)K);
  };
#else
  // fallback: cooperative global->LDS copy (b128 chunks)
  auto stage = [&](int buf, int k0) {
#pragma unroll
    for (int c = 0; c < 4; ++c) {            // A tile: 64 rows * 128B
      int id = threadIdx.x + 128 * c;
      int row = id >> 3, e = (id & 7) * 8;
      *(u32x4*)&smem[buf][row * KSTEP + e] =
          *(const u32x4*)(A + (size_t)(m0 + row) * K + k0 + e);
    }
#pragma unroll
    for (int c = 0; c < 8; ++c) {            // B tile: 128 rows * 128B
      int id = threadIdx.x + 128 * c;
      int row = id >> 3, e = (id & 7) * 8;
      *(u32x4*)&smem[buf][TILE_A_ELE + row * KSTEP + e] =
          *(const u32x4*)(Bt + (size_t)(n0 + row) * K + k0 + e);
    }
  };
#endif

  v8f c[2][4] = {};

  stage(0, 0);
#if USE_TDM
  wait_tensor0();
#endif
  __syncthreads();

  int cur = 0;
  for (int k0 = 0; k0 < K; k0 += KSTEP, cur ^= 1) {
    if (k0 + KSTEP < K) stage(cur ^ 1, k0 + KSTEP);  // DMA next during compute

    const bf16* sA = &smem[cur][0];
    const bf16* sB = &smem[cur][TILE_A_ELE];
#pragma unroll
    for (int c2 = 0; c2 < 2; ++c2) {         // two 32-K chunks of this stage
      v16bf a0 = load_frag_a(sA + (size_t)(wm + cn) * KSTEP + c2 * 32, lane);
      v16bf a1 = load_frag_a(sA + (size_t)(wm + 16 + cn) * KSTEP + c2 * 32, lane);
#pragma unroll
      for (int j = 0; j < 4; ++j) {
        v16bf bb =
            load_frag_b(sB + (size_t)(wn + 16 * j + cn) * KSTEP + c2 * 32, lane);
        c[0][j] = wmma_bf16(a0, bb, c[0][j]);
        c[1][j] = wmma_bf16(a1, bb, c[1][j]);
      }
    }

#if USE_TDM
    wait_tensor0();                          // own TDM slice for next buffer
#endif
    __syncthreads();                         // all slices landed; cur free
  }

  // C fragment: lane l, elem r -> row r + 8*(l>=16), col l&15 (ISA 7.12.2)
  const int rbase = (lane & 16) ? 8 : 0;
#pragma unroll
  for (int i = 0; i < 2; ++i)
#pragma unroll
    for (int j = 0; j < 4; ++j)
#pragma unroll
      for (int r = 0; r < 8; ++r) {
        size_t idx = (size_t)(m0 + wm + 16 * i + rbase + r) * N +
                     (n0 + wn + 16 * j + cn);
        if (OUT_F32) ((float*)Cout)[idx] = c[i][j][r];
        else         ((bf16*)Cout)[idx]  = f2bf(c[i][j][r]);
      }
}

// ---------------------------------------------------------------------------
// RoPE + head transpose: proj[B*S, H*128] -> outT[B,H,S,128]
// ---------------------------------------------------------------------------
__global__ void k_rope(const bf16* __restrict__ proj,
                       const float* __restrict__ freqs,
                       bf16* __restrict__ outT, int H, size_t total) {
  size_t idx = (size_t)blockIdx.x * blockDim.x + threadIdx.x;
  if (idx >= total) return;                  // total = B*S*H*64
  int i  = (int)(idx & 63);
  int hh = (int)((idx >> 6) % H);
  int s  = (int)(((idx >> 6) / H) % Sn);
  int b  = (int)(idx / ((size_t)64 * H * Sn));
  float f = freqs[s * 64 + i];
  float cs = cosf(f), sn = sinf(f);
  size_t src = ((size_t)(b * Sn + s) * H + hh) * HD + 2 * i;
  float re = bf2f(proj[src]), im = bf2f(proj[src + 1]);
  size_t dst = (((size_t)(b * H + hh) * Sn) + s) * HD + 2 * i;
  outT[dst]     = f2bf(re * cs - im * sn);
  outT[dst + 1] = f2bf(re * sn + im * cs);
}

// vproj[B*S, HKV*128] -> vT[B,HKV,128,S]  (key dim contiguous for B frags)
__global__ void k_transpose_v(const bf16* __restrict__ vproj,
                              bf16* __restrict__ vT, size_t total) {
  size_t idx = (size_t)blockIdx.x * blockDim.x + threadIdx.x;
  if (idx >= total) return;                  // total = B*S*NKV
  int d  = (int)(idx & 127);
  int hh = (int)((idx >> 7) & 7);
  int s  = (int)((idx >> 10) % Sn);
  int b  = (int)(idx / ((size_t)NKV * Sn));
  vT[(((size_t)(b * HKV + hh) * HD) + d) * Sn + s] = vproj[idx];
}

// ---------------------------------------------------------------------------
// Causal flash attention (GQA).  Block = 4 waves; each wave owns 16 q-rows x
// full 128 hd.  Per 32-key tile: 8 WMMA (scores) + online softmax (shfl-xor
// over 16-lane row groups) + per-wave LDS transpose of P + 8 WMMA (P*V).
// ---------------------------------------------------------------------------
__global__ void __launch_bounds__(128)
k_attention(const bf16* __restrict__ qT, const bf16* __restrict__ kT,
            const bf16* __restrict__ vT, bf16* __restrict__ attn) {
  __shared__ bf16 ldsP[4][16 * 32];          // per-wave 16x32 P tile

  const int lane  = threadIdx.x & 31;
  const int w     = threadIdx.x >> 5;
  const int b     = blockIdx.z;
  const int h     = blockIdx.y;
  const int kvh   = h >> 2;                  // GROUPS = 4
  const int qbase = blockIdx.x * 64 + w * 16;
  const int rbase = (lane & 16) ? 8 : 0;
  const int cn    = lane & 15;
  const float scale = 0.08838834764831843f;  // 1/sqrt(128)

  const bf16* qrow = qT + ((size_t)(b * HQ + h) * Sn + qbase + cn) * HD;
  v16bf qa[4];
#pragma unroll
  for (int kk = 0; kk < 4; ++kk) qa[kk] = load_frag_a(qrow + kk * 32, lane);

  const bf16* kbase = kT + (size_t)(b * HKV + kvh) * Sn * HD;
  const bf16* vbase = vT + (size_t)(b * HKV + kvh) * HD * Sn;

  v8f o[8] = {};
  float mr[8], lr[8];
#pragma unroll
  for (int r = 0; r < 8; ++r) { mr[r] = -1e30f; lr[r] = 0.0f; }

  for (int kt = 0; kt < qbase + 16; kt += 32) {
    if (kt + 32 < qbase + 16) {              // prefetch next key tile
      __builtin_prefetch(kbase + (size_t)(kt + 32 + cn) * HD, 0, 0);
      __builtin_prefetch(vbase + (size_t)cn * Sn + kt + 32, 0, 0);
    }

    // ---- scores: 16x32 tile = two 16x16 WMMA columns, K=128 (4 steps) ----
    v8f sc[2] = {};
#pragma unroll
    for (int c = 0; c < 2; ++c) {
      const bf16* krow = kbase + (size_t)(kt + c * 16 + cn) * HD;
#pragma unroll
      for (int kk = 0; kk < 4; ++kk)
        sc[c] = wmma_bf16(qa[kk], load_frag_b(krow + kk * 32, lane), sc[c]);
    }

    // ---- scale + causal mask ----
    float sv[2][8];
#pragma unroll
    for (int c = 0; c < 2; ++c)
#pragma unroll
      for (int r = 0; r < 8; ++r) {
        int key = kt + c * 16 + cn;
        int qr  = qbase + rbase + r;
        float x = sc[c][r] * scale;
        sv[c][r] = (key <= qr) ? x : -1e30f;
      }

    // ---- online softmax; rows live in 16-lane groups (C-frag layout) ----
#pragma unroll
    for (int r = 0; r < 8; ++r) {
      float rm = fmaxf(sv[0][r], sv[1][r]);
      rm = fmaxf(rm, __shfl_xor(rm, 1, 32));
      rm = fmaxf(rm, __shfl_xor(rm, 2, 32));
      rm = fmaxf(rm, __shfl_xor(rm, 4, 32));
      rm = fmaxf(rm, __shfl_xor(rm, 8, 32));
      float nm  = fmaxf(mr[r], rm);
      float fac = __expf(mr[r] - nm);
      mr[r] = nm;
      float p0 = __expf(sv[0][r] - nm);
      float p1 = __expf(sv[1][r] - nm);
      float rs = p0 + p1;
      rs += __shfl_xor(rs, 1, 32);
      rs += __shfl_xor(rs, 2, 32);
      rs += __shfl_xor(rs, 4, 32);
      rs += __shfl_xor(rs, 8, 32);
      lr[r] = lr[r] * fac + rs;
      int m = rbase + r;
      ldsP[w][m * 32 + cn]      = f2bf(p0);
      ldsP[w][m * 32 + 16 + cn] = f2bf(p1);
#pragma unroll
      for (int n = 0; n < 8; ++n) o[n][r] *= fac;
    }

    // ---- P (C-layout f32) -> A fragment via per-wave LDS bounce ----
    v16bf pa = load_frag_a(&ldsP[w][cn * 32], lane);

    // ---- O += P * V : 8 hd chunks of 16 ----
#pragma unroll
    for (int n = 0; n < 8; ++n) {
      const bf16* vrow = vbase + (size_t)(n * 16 + cn) * Sn + kt;
      o[n] = wmma_bf16(pa, load_frag_b(vrow, lane), o[n]);
    }
  }

  // ---- normalize & write attn output [B*S, HQ*128] ----
#pragma unroll
  for (int n = 0; n < 8; ++n)
#pragma unroll
    for (int r = 0; r < 8; ++r) {
      int row = qbase + rbase + r;
      size_t dst = ((size_t)b * Sn + row) * NQ + h * HD + n * 16 + cn;
      attn[dst] = f2bf(o[n][r] / lr[r]);
    }
}

// ---------------------------------------------------------------------------
// Host launch
// ---------------------------------------------------------------------------
extern "C" void kernel_launch(void* const* d_in, const int* in_sizes, int n_in,
                              void* d_out, int out_size, void* d_ws,
                              size_t ws_size, hipStream_t stream) {
  const float* x     = (const float*)d_in[0];
  const float* freqs = (const float*)d_in[1];
  // d_in[2] = additive causal mask (recomputed analytically in-kernel)
  const float* wq = (const float*)d_in[3];
  const float* wk = (const float*)d_in[4];
  const float* wv = (const float*)d_in[5];
  const float* wo = (const float*)d_in[6];
  float* out = (float*)d_out;

  char* ws = (char*)d_ws;
  size_t off = 0;
  bf16* xb    = (bf16*)(ws + off); off += (size_t)MROW * Dn * 2;   // reused as qT
  bf16* wqT   = (bf16*)(ws + off); off += (size_t)NQ  * Dn * 2;
  bf16* wkT   = (bf16*)(ws + off); off += (size_t)NKV * Dn * 2;
  bf16* wvT   = (bf16*)(ws + off); off += (size_t)NKV * Dn * 2;
  bf16* woT   = (bf16*)(ws + off); off += (size_t)Dn  * Dn * 2;
  bf16* qproj = (bf16*)(ws + off); off += (size_t)MROW * NQ * 2;   // reused as attn
  bf16* kproj = (bf16*)(ws + off); off += (size_t)MROW * NKV * 2;
  bf16* vproj = (bf16*)(ws + off); off += (size_t)MROW * NKV * 2;
  bf16* kTb   = (bf16*)(ws + off); off += (size_t)MROW * NKV * 2;
  bf16* vTb   = (bf16*)(ws + off); off += (size_t)MROW * NKV * 2;
  bf16* qTb  = xb;      // x dead after the three QKV GEMMs
  bf16* attn = qproj;   // qproj dead after RoPE

  // 1) precision conversion (+ weight transpose to N-major)
  k_f32_to_bf16<<<4096, 256, 0, stream>>>(x, xb, (size_t)MROW * Dn);
  k_transpose_to_bf16<<<4096, 256, 0, stream>>>(wq, wqT, Dn, NQ);
  k_transpose_to_bf16<<<4096, 256, 0, stream>>>(wk, wkT, Dn, NKV);
  k_transpose_to_bf16<<<4096, 256, 0, stream>>>(wv, wvT, Dn, NKV);
  k_transpose_to_bf16<<<4096, 256, 0, stream>>>(wo, woT, Dn, Dn);

  // 2) QKV projections (bf16 out)
  k_gemm_bf16<0><<<dim3(NQ / 128,  MROW / 64), 128, 0, stream>>>(xb, wqT, qproj, MROW, NQ,  Dn);
  k_gemm_bf16<0><<<dim3(NKV / 128, MROW / 64), 128, 0, stream>>>(xb, wkT, kproj, MROW, NKV, Dn);
  k_gemm_bf16<0><<<dim3(NKV / 128, MROW / 64), 128, 0, stream>>>(xb, wvT, vproj, MROW, NKV, Dn);

  // 3) RoPE(Q,K) + head transposes
  {
    size_t tq = (size_t)Bn * Sn * HQ * 64;
    size_t tk = (size_t)Bn * Sn * HKV * 64;
    size_t tv = (size_t)Bn * Sn * NKV;
    k_rope<<<(unsigned)((tq + 255) / 256), 256, 0, stream>>>(qproj, freqs, qTb, HQ, tq);
    k_rope<<<(unsigned)((tk + 255) / 256), 256, 0, stream>>>(kproj, freqs, kTb, HKV, tk);
    k_transpose_v<<<(unsigned)((tv + 255) / 256), 256, 0, stream>>>(vproj, vTb, tv);
  }

  // 4) causal flash attention
  k_attention<<<dim3(Sn / 64, HQ, Bn), 128, 0, stream>>>(qTb, kTb, vTb, attn);

  // 5) output projection (f32 out)
  k_gemm_bf16<1><<<dim3(Dn / 128, MROW / 64), 128, 0, stream>>>(attn, woT, out, MROW, Dn, Dn);
}
